// CorrelationModule_70755291235113
// MI455X (gfx1250) — matrix operand order repaired and checked by reading, not compiled, not run on hardware
//
#include <hip/hip_runtime.h>

// ---------------------------------------------------------------------------
// CDNA5 (gfx1250) implementation of the correlation module.
// All matrix math routed through v_wmma_f32_16x16x32_f16 (wave32 WMMA).
// Activations are NHWC f16 with zero-padded halo columns ([N][H][W+2][C]) so
// the WMMA B-fragment loads need no per-lane boundary branches.
// Each wave computes a 32-cout x 16-pixel tile (2 A fragments share 1 B).
// ---------------------------------------------------------------------------

typedef __attribute__((ext_vector_type(16))) _Float16 v16h;
typedef __attribute__((ext_vector_type(8)))  float    v8f;
typedef __attribute__((ext_vector_type(4)))  float    f4;

union H16 { v16h h; f4 f[2]; };          // 16 halves = one WMMA A/B fragment
struct h32 { f4 q[4]; };                 // 32 halves (64B)

__device__ __forceinline__ f4 ldg16(const _Float16* p) { return *(const f4*)p; }

// ---------------------------------------------------------------------------
// Generic 3x3 implicit-GEMM conv, pad=1, stride in {1,2}.
// act: [N][Hin][Wpin][Cin] f16 halo-padded NHWC, wgt: [3][3][Cout][Cin] f16.
// Optional: residual add (f32 padded NHWC, image index n/rdiv), f16 raw out,
// f32 out, fused per-channel sum/sumsq atomics for training-mode BN stats.
// ---------------------------------------------------------------------------
__global__ __launch_bounds__(128) void conv3x3_wmma(
    const _Float16* __restrict__ act,
    const _Float16* __restrict__ wgt,
    const float*    __restrict__ resid,
    _Float16*       __restrict__ out_h,
    float*          __restrict__ out_f,
    float*          __restrict__ ssum,
    float*          __restrict__ ssq,
    int N, int Hin, int Wpin, int Cin,
    int Hout, int Wout, int Wpout, int Cout, int stride, int rdiv)
{
    const int lane = threadIdx.x & 31;
    const int hi = lane >> 4, lo = lane & 15;
    int wid = blockIdx.x * (blockDim.x >> 5) + (threadIdx.x >> 5);
    const int ctiles = Cout >> 5;                   // 32-cout tiles
    const int wtiles = Wout >> 4;
    const long total = (long)ctiles * N * Hout * wtiles;
    if (wid >= total) return;                       // wave-uniform exit

    int ct = wid % ctiles; int pt = wid / ctiles;
    int wt = pt % wtiles;  pt /= wtiles;
    int h  = pt % Hout;    int n = pt / Hout;
    int w0 = wt << 4;

    v8f acc0 = {0.f,0.f,0.f,0.f,0.f,0.f,0.f,0.f};
    v8f acc1 = {0.f,0.f,0.f,0.f,0.f,0.f,0.f,0.f};
    const int xb = (w0 + lo) * stride;              // lane's input column base
    const _Float16* wct = wgt + (size_t)(ct * 32 + lo) * Cin;

    for (int kh = 0; kh < 3; ++kh) {
        int ih = h * stride + kh - 1;
        if (ih < 0 || ih >= Hin) continue;          // uniform across wave
        const _Float16* arow = act + ((size_t)n * Hin + ih) * (size_t)Wpin * Cin;
        for (int kw = 0; kw < 3; ++kw) {
            // halo-padded: logical iw = xb+kw-1 -> physical column iw+1 >= 0
            const _Float16* brow = arow + (size_t)(xb + kw) * Cin;
            const _Float16* wrow0 = wct + (size_t)((kh * 3 + kw) * Cout) * Cin;
            const _Float16* wrow1 = wrow0 + (size_t)16 * Cin;
            for (int c0 = 0; c0 < Cin; c0 += 32) {
                H16 a0, a1, b;
                // A fragments (16x32: M=lo, K split 8+8 by lane half)
                a0.f[0] = ldg16(wrow0 + c0 + hi * 8);
                a0.f[1] = ldg16(wrow0 + c0 + 16 + hi * 8);
                a1.f[0] = ldg16(wrow1 + c0 + hi * 8);
                a1.f[1] = ldg16(wrow1 + c0 + 16 + hi * 8);
                // shared B fragment (32x16: N=lo, K split 8+8 by lane half)
                b.f[0] = ldg16(brow + c0 + hi * 8);
                b.f[1] = ldg16(brow + c0 + 16 + hi * 8);
                acc0 = __builtin_amdgcn_wmma_f32_16x16x32_f16(
                         false, a0.h, false, b.h, (short)0, acc0, false, false);
                acc1 = __builtin_amdgcn_wmma_f32_16x16x32_f16(
                         false, a1.h, false, b.h, (short)0, acc1, false, false);
            }
        }
    }

    const int wpix = w0 + lo;
    const size_t obase = (((size_t)n * Hout + h) * Wpout + (wpix + 1)) * Cout + ct * 32;
    const size_t rbase = resid
        ? ((((size_t)(n / rdiv) * Hout + h) * Wpout + (wpix + 1)) * Cout + ct * 32) : 0;
#pragma unroll
    for (int f = 0; f < 2; ++f) {
        v8f acc = f ? acc1 : acc0;
#pragma unroll
        for (int r = 0; r < 8; ++r) {
            int M = f * 16 + r + hi * 8;
            float val = acc[r];
            if (resid) val += resid[rbase + M];
            if (out_f) out_f[obase + M] = val;
            if (out_h) out_h[obase + M] = (_Float16)val;
            if (ssum) {
                float s = val, q = val * val;
#pragma unroll
                for (int m = 8; m >= 1; m >>= 1) {  // reduce 16 pixel lanes
                    s += __shfl_xor(s, m, 32);
                    q += __shfl_xor(q, m, 32);
                }
                if (lo == 0) {
                    atomicAdd(&ssum[ct * 32 + M], s);
                    atomicAdd(&ssq [ct * 32 + M], q);
                }
            }
        }
    }
}

// ---------------------------------------------------------------------------
// ConvTranspose2d 4x4, stride 2, pad 1 (64 -> 32 ch), via per-output-parity
// 2x2-tap WMMA GEMMs (taps are wave-uniform within a parity class).
// act: [N][Hin][Wpin][64] halo-padded, wgt: [4][4][32][64],
// out: [N][2Hin][2Win+2][32] (halo-padded).
// ---------------------------------------------------------------------------
__global__ __launch_bounds__(128) void deconv4x4s2_wmma(
    const _Float16* __restrict__ act,
    const _Float16* __restrict__ wgt,
    _Float16*       __restrict__ out_h,
    float* __restrict__ ssum, float* __restrict__ ssq,
    int N, int Hin, int Win, int Wpin, int Wpout)
{
    const int Cin = 64, Cout = 32;
    const int lane = threadIdx.x & 31;
    const int hi = lane >> 4, lo = lane & 15;
    int wid = blockIdx.x * (blockDim.x >> 5) + (threadIdx.x >> 5);
    const int wtiles = Win >> 4;
    const long total = (long)N * 4 * Hin * wtiles;
    if (wid >= total) return;

    int wt = wid % wtiles; int pt = wid / wtiles;
    int y  = pt % Hin;     pt /= Hin;
    int pw = pt & 1;       pt >>= 1;
    int ph = pt & 1;       int n = pt >> 1;
    int x  = (wt << 4) + lo;

    int khs[2], dih[2], kws[2], diw[2];
    if (ph == 0) { khs[0] = 1; dih[0] = 0; khs[1] = 3; dih[1] = -1; }
    else         { khs[0] = 0; dih[0] = 1; khs[1] = 2; dih[1] =  0; }
    if (pw == 0) { kws[0] = 1; diw[0] = 0; kws[1] = 3; diw[1] = -1; }
    else         { kws[0] = 0; diw[0] = 1; kws[1] = 2; diw[1] =  0; }

    v8f acc0 = {0.f,0.f,0.f,0.f,0.f,0.f,0.f,0.f};
    v8f acc1 = {0.f,0.f,0.f,0.f,0.f,0.f,0.f,0.f};
    for (int th = 0; th < 2; ++th) {
        int ih = y + dih[th];
        if (ih < 0 || ih >= Hin) continue;
        const _Float16* arow = act + ((size_t)n * Hin + ih) * (size_t)Wpin * Cin;
        for (int tw = 0; tw < 2; ++tw) {
            const _Float16* brow = arow + (size_t)(x + diw[tw] + 1) * Cin;
            const _Float16* wrow0 = wgt
                + (size_t)((khs[th] * 4 + kws[tw]) * Cout + lo) * Cin;
            const _Float16* wrow1 = wrow0 + (size_t)16 * Cin;
            for (int c0 = 0; c0 < Cin; c0 += 32) {
                H16 a0, a1, b;
                a0.f[0] = ldg16(wrow0 + c0 + hi * 8);
                a0.f[1] = ldg16(wrow0 + c0 + 16 + hi * 8);
                a1.f[0] = ldg16(wrow1 + c0 + hi * 8);
                a1.f[1] = ldg16(wrow1 + c0 + 16 + hi * 8);
                b.f[0]  = ldg16(brow + c0 + hi * 8);
                b.f[1]  = ldg16(brow + c0 + 16 + hi * 8);
                acc0 = __builtin_amdgcn_wmma_f32_16x16x32_f16(
                         false, a0.h, false, b.h, (short)0, acc0, false, false);
                acc1 = __builtin_amdgcn_wmma_f32_16x16x32_f16(
                         false, a1.h, false, b.h, (short)0, acc1, false, false);
            }
        }
    }

    const int oh = 2 * y + ph, ow = 2 * x + pw;
    const int Hout = 2 * Hin;
    const size_t obase = (((size_t)n * Hout + oh) * Wpout + (ow + 1)) * Cout;
#pragma unroll
    for (int f = 0; f < 2; ++f) {
        v8f acc = f ? acc1 : acc0;
#pragma unroll
        for (int r = 0; r < 8; ++r) {
            int M = f * 16 + r + hi * 8;
            float val = acc[r];
            out_h[obase + M] = (_Float16)val;
            float s = val, q = val * val;
#pragma unroll
            for (int m = 8; m >= 1; m >>= 1) {
                s += __shfl_xor(s, m, 32);
                q += __shfl_xor(q, m, 32);
            }
            if (lo == 0) {
                atomicAdd(&ssum[M], s);
                atomicAdd(&ssq [M], q);
            }
        }
    }
}

// ---------------------------------------------------------------------------
// Small helper kernels
// ---------------------------------------------------------------------------
__global__ void zero_f(float* p, int n) {
    int i = blockIdx.x * blockDim.x + threadIdx.x;
    if (i < n) p[i] = 0.f;
}

// Zero the two halo columns of a padded NHWC f16 tensor [N][H][Wp][C]
__global__ void zero_halo(_Float16* __restrict__ x, int N, int H, int Wp, int C)
{
    long gid = (long)blockIdx.x * blockDim.x + threadIdx.x;
    long total = (long)N * H * 2 * C;
    if (gid >= total) return;
    int c = gid % C; long t = gid / C;
    int side = t & 1;  t >>= 1;
    int h = t % H;     int n = t / H;
    int w = side ? (Wp - 1) : 0;
    x[(((size_t)n * H + h) * Wp + w) * C + c] = (_Float16)0.f;
}

// NCHW f32 -> NHWC f16 (optionally halo-padded: write at column w+pad)
__global__ void nchw2nhwc_h(const float* __restrict__ src,
                            _Float16* __restrict__ dst,
                            int N, int C, int H, int W, int Wp, int pad)
{
    long gid = (long)blockIdx.x * blockDim.x + threadIdx.x;
    long total = (long)N * C * H * W;
    if (gid >= total) return;
    int w = gid % W; long t = gid / W;
    int h = t % H;   t /= H;
    int c = t % C;   int n = t / C;
    dst[(((size_t)n * H + h) * Wp + w + pad) * C + c] = (_Float16)src[gid];
}

// OIHW f32 -> [kh][kw][Cout][Cin] f16 (with channel-offset slicing for w1 split)
__global__ void prep_w3(const float* __restrict__ src, _Float16* __restrict__ dst,
                        int Cout, int Cin, int CinFull, int coff)
{
    long gid = (long)blockIdx.x * blockDim.x + threadIdx.x;
    long total = (long)9 * Cout * Cin;
    if (gid >= total) return;
    int i = gid % Cin; long t = gid / Cin;
    int o = t % Cout;  int k = t / Cout;      // k = kh*3+kw
    int kh = k / 3, kw = k % 3;
    dst[gid] = (_Float16)src[(((size_t)o * CinFull + coff + i) * 3 + kh) * 3 + kw];
}

// torch ConvTranspose (I=64,O=32,4,4) f32 -> [kh][kw][32][64] f16
__global__ void prep_wd(const float* __restrict__ src, _Float16* __restrict__ dst)
{
    int gid = blockIdx.x * blockDim.x + threadIdx.x;
    if (gid >= 16 * 32 * 64) return;
    int i = gid % 64; int t = gid / 64;
    int o = t % 32;   int k = t / 32;
    int kh = k / 4, kw = k % 4;
    dst[gid] = (_Float16)src[(((size_t)i * 32 + o) * 4 + kh) * 4 + kw];
}

// Bilinear sample of f2 (unpadded NHWC f16) at coords + displacement grid
// -> f2s (halo-padded NHWC f16). One thread = 32 channels of one pixel.
__global__ void sample_corr(const _Float16* __restrict__ f2h,
                            const float* __restrict__ coords,
                            _Float16* __restrict__ f2s,
                            int Bt, int H, int W, int Wp, int C)
{
    long gid = (long)blockIdx.x * blockDim.x + threadIdx.x;
    const int chunks = C / 32;
    long total = (long)Bt * 49 * H * W * chunks;
    if (gid >= total) return;
    int cc = gid % chunks; long p = gid / chunks;
    int w = p % W; p /= W;
    int h = p % H; p /= H;
    int d = p % 49; int b = p / 49;
    int ix = d / 7, iy = d % 7;                   // x-offset major (matches ref)

    float gx = coords[(((size_t)b * 2 + 0) * H + h) * W + w] + (float)(ix - 3);
    float gy = coords[(((size_t)b * 2 + 1) * H + h) * W + w] + (float)(iy - 3);
    float x0f = floorf(gx), y0f = floorf(gy);
    float wx = gx - x0f, wy = gy - y0f;
    int x0 = (int)x0f, y0 = (int)y0f;
    float w00 = (1.f - wx) * (1.f - wy), w10 = wx * (1.f - wy);
    float w01 = (1.f - wx) * wy,         w11 = wx * wy;

    const _Float16* base = f2h + (size_t)b * H * W * C + cc * 32;
    float acc[32];
#pragma unroll
    for (int i = 0; i < 32; ++i) acc[i] = 0.f;

    auto corner = [&](int xi, int yi, float wt) {
        if (xi >= 0 && xi < W && yi >= 0 && yi < H) {
            h32 t = *(const h32*)(base + ((size_t)yi * W + xi) * C);
            const _Float16* th = (const _Float16*)&t;
#pragma unroll
            for (int i = 0; i < 32; ++i) acc[i] += wt * (float)th[i];
        }
    };
    corner(x0,     y0,     w00);
    corner(x0 + 1, y0,     w10);
    corner(x0,     y0 + 1, w01);
    corner(x0 + 1, y0 + 1, w11);

    _Float16* out = f2s + ((((size_t)b * 49 + d) * H + h) * Wp + (w + 1)) * C + cc * 32;
    h32 o;
    _Float16* oh = (_Float16*)&o;
#pragma unroll
    for (int i = 0; i < 32; ++i) oh[i] = (_Float16)acc[i];
    *(h32*)out = o;
}

// BN finalize: mul = g*rsqrt(var+eps), add = b - mu*mul  (biased variance)
__global__ void bn_fin(const float* __restrict__ sum, const float* __restrict__ sq,
                       const float* __restrict__ g, const float* __restrict__ b,
                       float* __restrict__ mul, float* __restrict__ add,
                       int C, float inv_count)
{
    int c = blockIdx.x * blockDim.x + threadIdx.x;
    if (c >= C) return;
    float mu  = sum[c] * inv_count;
    float var = sq[c] * inv_count - mu * mu;
    float m   = g[c] * rsqrtf(var + 1e-5f);
    mul[c] = m;
    add[c] = b[c] - mu * m;
}

// In-place BN + ReLU over halo-padded NHWC f16; halo columns forced to zero
// so the tensor remains a valid zero-padded conv input. 8 halves / thread.
__global__ void bn_apply(_Float16* __restrict__ x,
                         const float* __restrict__ mul, const float* __restrict__ add,
                         long total, int C, int Wp, int Wv)
{
    long gid = (long)blockIdx.x * blockDim.x + threadIdx.x;
    long i8 = gid * 8;
    if (i8 >= total) return;
    union { f4 f; _Float16 h[8]; } u;
    u.f = *(f4*)(x + i8);
    long pix = i8 / C;
    int w = (int)(pix % Wp);
    bool valid = (w >= 1) && (w <= Wv);
    int c0 = (int)(i8 % C);
#pragma unroll
    for (int j = 0; j < 8; ++j) {
        float v = (float)u.h[j] * mul[c0 + j] + add[c0 + j];
        u.h[j] = valid ? (_Float16)fmaxf(v, 0.f) : (_Float16)0.f;
    }
    *(f4*)(x + i8) = u.f;
}

// Final 3x3 conv, 32 -> 1 channel, weights cached in LDS. a5 halo-padded.
__global__ void conv6_k(const _Float16* __restrict__ a5,
                        const float* __restrict__ w6, const float* __restrict__ b6,
                        float* __restrict__ cost, int N, int H, int W, int Wp)
{
    __shared__ float sw[288];
    for (int i = threadIdx.x; i < 288; i += blockDim.x) sw[i] = w6[i];
    __syncthreads();
    long gid = (long)blockIdx.x * blockDim.x + threadIdx.x;
    long total = (long)N * H * W;
    if (gid >= total) return;
    int w = gid % W; long t = gid / W;
    int h = t % H;   int n = t / H;
    float s = b6[0];
    for (int kh = 0; kh < 3; ++kh) {
        int ih = h + kh - 1;
        if (ih < 0 || ih >= H) continue;
        for (int kw = 0; kw < 3; ++kw) {
            const _Float16* p = a5 + (((size_t)n * H + ih) * Wp + (w + kw)) * 32;
#pragma unroll
            for (int c = 0; c < 32; ++c)
                s += (float)p[c] * sw[c * 9 + kh * 3 + kw];
        }
    }
    cost[gid] = s;
}

// DAP: out[b,e,h,w] = sum_d dap[e,d] * cost[b*49+d,h,w]; dap cached in LDS.
__global__ void dap_k(const float* __restrict__ cost, const float* __restrict__ dap,
                      float* __restrict__ out, int Bt, int H, int W)
{
    __shared__ float sd[2401];
    for (int i = threadIdx.x; i < 2401; i += blockDim.x) sd[i] = dap[i];
    __syncthreads();
    long gid = (long)blockIdx.x * blockDim.x + threadIdx.x;
    long total = (long)Bt * 49 * H * W;
    if (gid >= total) return;
    int w = gid % W; long t = gid / W;
    int h = t % H;   t /= H;
    int e = t % 49;  int b = t / 49;
    float s = 0.f;
#pragma unroll 7
    for (int d = 0; d < 49; ++d)
        s += sd[e * 49 + d] * cost[(((size_t)(b * 49 + d)) * H + h) * W + w];
    out[gid] = s;
}

// ---------------------------------------------------------------------------
// Host-side orchestration
// ---------------------------------------------------------------------------
extern "C" void kernel_launch(void* const* d_in, const int* in_sizes, int n_in,
                              void* d_out, int out_size, void* d_ws, size_t ws_size,
                              hipStream_t stream)
{
    const float* f1     = (const float*)d_in[0];
    const float* f2     = (const float*)d_in[1];
    const float* coords = (const float*)d_in[2];
    const float* w1 = (const float*)d_in[3];
    const float* g1 = (const float*)d_in[4];  const float* b1 = (const float*)d_in[5];
    const float* w2 = (const float*)d_in[6];
    const float* g2 = (const float*)d_in[7];  const float* b2 = (const float*)d_in[8];
    const float* w3 = (const float*)d_in[9];
    const float* g3 = (const float*)d_in[10]; const float* b3 = (const float*)d_in[11];
    const float* w4 = (const float*)d_in[12];
    const float* g4 = (const float*)d_in[13]; const float* b4 = (const float*)d_in[14];
    const float* w5 = (const float*)d_in[15];
    const float* g5 = (const float*)d_in[16]; const float* b5 = (const float*)d_in[17];
    const float* w6 = (const float*)d_in[18]; const float* b6 = (const float*)d_in[19];
    const float* dapw = (const float*)d_in[20];
    float* out = (float*)d_out;

    const int B = 2, C = 128, H = 48, W = 96, NB = 98;      // NB = B*49
    const int H2 = 24, W2 = 48;
    const int Wp = W + 2, Wp2 = W2 + 2;                     // halo-padded widths

    // bump allocator over workspace
    char* wp = (char*)d_ws;
    auto alloc = [&](size_t bytes) -> char* {
        char* p = wp; wp += (bytes + 255) & ~(size_t)255; return p;
    };
    _Float16* f1h  = (_Float16*)alloc((size_t)B * H * Wp * C * 2);   // padded
    _Float16* f2h  = (_Float16*)alloc((size_t)B * H * W * C * 2);    // unpadded (gather src)
    _Float16* f2s  = (_Float16*)alloc((size_t)NB * H * Wp * C * 2);  // padded
    _Float16* w1a  = (_Float16*)alloc((size_t)9 * 96 * 128 * 2);
    _Float16* w1b  = (_Float16*)alloc((size_t)9 * 96 * 128 * 2);
    _Float16* w2p  = (_Float16*)alloc((size_t)9 * 128 * 96 * 2);
    _Float16* w3p  = (_Float16*)alloc((size_t)9 * 128 * 128 * 2);
    _Float16* w4p  = (_Float16*)alloc((size_t)9 * 64 * 128 * 2);
    _Float16* w5p  = (_Float16*)alloc((size_t)16 * 32 * 64 * 2);
    float*    base1= (float*)   alloc((size_t)B * H * Wp * 96 * 4);  // padded stride
    _Float16* a1   = (_Float16*)alloc((size_t)NB * H * Wp * 96 * 2);
    _Float16* a2   = (_Float16*)alloc((size_t)NB * H2 * Wp2 * 128 * 2);
    _Float16* a3   = (_Float16*)alloc((size_t)NB * H2 * Wp2 * 128 * 2);
    _Float16* a4   = (_Float16*)alloc((size_t)NB * H2 * Wp2 * 64 * 2);
    _Float16* a5   = (_Float16*)alloc((size_t)NB * H * Wp * 32 * 2);
    float*    cost = (float*)   alloc((size_t)NB * H * W * 4);
    float*    stats= (float*)   alloc((size_t)5 * 512 * 4);
    // per layer l: sum = stats+l*512, sq = +128, mul = +256, add = +384

    auto blocks = [](long threads, int bs) { return (int)((threads + bs - 1) / bs); };

    // 0) zero BN accumulators + halos of non-BN conv inputs (deterministic)
    zero_f<<<blocks(2560, 256), 256, 0, stream>>>(stats, 2560);
    zero_halo<<<blocks((long)B * H * 2 * C, 256), 256, 0, stream>>>(f1h, B, H, Wp, C);
    zero_halo<<<blocks((long)NB * H * 2 * C, 256), 256, 0, stream>>>(f2s, NB, H, Wp, C);

    // 1) pack f1 (padded) / f2 (unpadded) to NHWC f16
    nchw2nhwc_h<<<blocks((long)B * C * H * W, 256), 256, 0, stream>>>(f1, f1h, B, C, H, W, Wp, 1);
    nchw2nhwc_h<<<blocks((long)B * C * H * W, 256), 256, 0, stream>>>(f2, f2h, B, C, H, W, W, 0);

    // 2) prepack weights
    prep_w3<<<blocks((long)9 * 96 * 128, 256), 256, 0, stream>>>(w1, w1a, 96, 128, 256, 0);
    prep_w3<<<blocks((long)9 * 96 * 128, 256), 256, 0, stream>>>(w1, w1b, 96, 128, 256, 128);
    prep_w3<<<blocks((long)9 * 128 * 96, 256), 256, 0, stream>>>(w2, w2p, 128, 96, 96, 0);
    prep_w3<<<blocks((long)9 * 128 * 128, 256), 256, 0, stream>>>(w3, w3p, 128, 128, 128, 0);
    prep_w3<<<blocks((long)9 * 64 * 128, 256), 256, 0, stream>>>(w4, w4p, 64, 128, 128, 0);
    prep_wd<<<blocks(16 * 32 * 64, 256), 256, 0, stream>>>(w5, w5p);

    // 3) conv1a: f1 half of conv1 (shared across the 49 displacements) -> f32 base
    {
        long waves = (long)(96 / 32) * B * H * (W / 16);
        conv3x3_wmma<<<blocks(waves, 4), 128, 0, stream>>>(
            f1h, w1a, nullptr, nullptr, base1, nullptr, nullptr,
            B, H, Wp, 128, H, W, Wp, 96, 1, 1);
    }

    // 4) bilinear-sampled cost-volume half (f2 at coords + delta)
    sample_corr<<<blocks((long)NB * H * W * 4, 256), 256, 0, stream>>>(
        f2h, coords, f2s, B, H, W, Wp, 128);

    // 5) conv1 = conv1b(f2s) + base1, fused BN stats (layer 0)
    {
        long waves = (long)(96 / 32) * NB * H * (W / 16);
        conv3x3_wmma<<<blocks(waves, 4), 128, 0, stream>>>(
            f2s, w1b, base1, a1, nullptr, stats + 0, stats + 128,
            NB, H, Wp, 128, H, W, Wp, 96, 1, 49);
    }
    bn_fin<<<1, 128, 0, stream>>>(stats, stats + 128, g1, b1, stats + 256, stats + 384,
                                  96, 1.f / (float)((long)NB * H * W));
    {
        long tot = (long)NB * H * Wp * 96;
        bn_apply<<<blocks(tot / 8, 256), 256, 0, stream>>>(a1, stats + 256, stats + 384,
                                                           tot, 96, Wp, W);
    }

    // 6) conv2 (96 -> 128, stride 2) + BN (layer 1)
    {
        long waves = (long)(128 / 32) * NB * H2 * (W2 / 16);
        conv3x3_wmma<<<blocks(waves, 4), 128, 0, stream>>>(
            a1, w2p, nullptr, a2, nullptr, stats + 512, stats + 640,
            NB, H, Wp, 96, H2, W2, Wp2, 128, 2, 1);
    }
    bn_fin<<<1, 128, 0, stream>>>(stats + 512, stats + 640, g2, b2,
                                  stats + 768, stats + 896,
                                  128, 1.f / (float)((long)NB * H2 * W2));
    {
        long tot = (long)NB * H2 * Wp2 * 128;
        bn_apply<<<blocks(tot / 8, 256), 256, 0, stream>>>(a2, stats + 768, stats + 896,
                                                           tot, 128, Wp2, W2);
    }

    // 7) conv3 (128 -> 128) + BN (layer 2)
    {
        long waves = (long)(128 / 32) * NB * H2 * (W2 / 16);
        conv3x3_wmma<<<blocks(waves, 4), 128, 0, stream>>>(
            a2, w3p, nullptr, a3, nullptr, stats + 1024, stats + 1152,
            NB, H2, Wp2, 128, H2, W2, Wp2, 128, 1, 1);
    }
    bn_fin<<<1, 128, 0, stream>>>(stats + 1024, stats + 1152, g3, b3,
                                  stats + 1280, stats + 1408,
                                  128, 1.f / (float)((long)NB * H2 * W2));
    {
        long tot = (long)NB * H2 * Wp2 * 128;
        bn_apply<<<blocks(tot / 8, 256), 256, 0, stream>>>(a3, stats + 1280, stats + 1408,
                                                           tot, 128, Wp2, W2);
    }

    // 8) conv4 (128 -> 64) + BN (layer 3)
    {
        long waves = (long)(64 / 32) * NB * H2 * (W2 / 16);
        conv3x3_wmma<<<blocks(waves, 4), 128, 0, stream>>>(
            a3, w4p, nullptr, a4, nullptr, stats + 1536, stats + 1664,
            NB, H2, Wp2, 128, H2, W2, Wp2, 64, 1, 1);
    }
    bn_fin<<<1, 128, 0, stream>>>(stats + 1536, stats + 1664, g4, b4,
                                  stats + 1792, stats + 1920,
                                  64, 1.f / (float)((long)NB * H2 * W2));
    {
        long tot = (long)NB * H2 * Wp2 * 64;
        bn_apply<<<blocks(tot / 8, 256), 256, 0, stream>>>(a4, stats + 1792, stats + 1920,
                                                           tot, 64, Wp2, W2);
    }

    // 9) deconv (64 -> 32, x2 upsample) + BN (layer 4)
    {
        long waves = (long)NB * 4 * H2 * (W2 / 16);
        deconv4x4s2_wmma<<<blocks(waves, 4), 128, 0, stream>>>(
            a4, w5p, a5, stats + 2048, stats + 2176, NB, H2, W2, Wp2, Wp);
    }
    bn_fin<<<1, 128, 0, stream>>>(stats + 2048, stats + 2176, g5, b5,
                                  stats + 2304, stats + 2432,
                                  32, 1.f / (float)((long)NB * H * W));
    {
        long tot = (long)NB * H * Wp * 32;
        bn_apply<<<blocks(tot / 8, 256), 256, 0, stream>>>(a5, stats + 2304, stats + 2432,
                                                           tot, 32, Wp, W);
    }

    // 10) conv6 (32 -> 1) + bias
    conv6_k<<<blocks((long)NB * H * W, 256), 256, 0, stream>>>(a5, w6, b6, cost, NB, H, W, Wp);

    // 11) DAP 49x49 mix
    dap_k<<<blocks((long)B * 49 * H * W, 256), 256, 0, stream>>>(cost, dapw, out, B, H, W);
}
